// SimpleBalancedModel_15968688406724
// MI455X (gfx1250) — matrix operand order repaired and checked by reading, not compiled, not run on hardware
//
#include <hip/hip_runtime.h>
#include <hip/hip_bf16.h>

#define N_NODES 100000
#define N_EDGES 1600000
#define N_GRAPH 256
#define DIM     128
#define HID     128
#define OUTD    2
#define STEPS   4

typedef __attribute__((ext_vector_type(16))) __bf16 v16bf;
typedef __attribute__((ext_vector_type(8)))  float  v8f;

union ABf { v16bf v; uint4 q[2]; };

__device__ __forceinline__ unsigned short f2bf(float f) {
    unsigned u = __float_as_uint(f);
    u += 0x7FFFu + ((u >> 16) & 1u);          // round to nearest even
    return (unsigned short)(u >> 16);
}
__device__ __forceinline__ float bf2f(unsigned h) {
    return __uint_as_float(h << 16);
}
__device__ __forceinline__ float sigmoidf_(float x) {
    return 1.0f / (1.0f + __expf(-x));
}

// A fragment: row-major A [M x 128] bf16. 16x32 tile at (row0, k0).
// lanes 0-15: row=lane, K {k0+0..7, k0+16..23}; lanes 16-31: K {k0+8..15, k0+24..31}
__device__ __forceinline__ v16bf load_a_frag(const unsigned short* A, int row0, int k0, int lane) {
    int r = lane & 15, hi = lane >> 4;
    const unsigned short* p = A + (row0 + r) * 128 + k0 + hi * 8;
    ABf u;
    u.q[0] = *(const uint4*)(p);
    u.q[1] = *(const uint4*)(p + 16);
    return u.v;
}

// B fragment: weight Wt [Ocols x 128] row-major (B[k][o] = Wt[o][k]). 32x16 tile at (k0, o0).
// lane: col = o0 + (lane&15); lanes 0-15: K {k0+0..7, k0+16..23}; lanes 16-31: K {k0+8..15, k0+24..31}
__device__ __forceinline__ v16bf load_b_frag(const unsigned short* Wt, int o0, int k0, int lane) {
    int c = lane & 15, hi = lane >> 4;
    const unsigned short* p = Wt + (o0 + c) * 128 + k0 + hi * 8;
    ABf u;
    u.q[0] = *(const uint4*)(p);
    u.q[1] = *(const uint4*)(p + 16);
    return u.v;
}

__device__ __forceinline__ v8f wmma_bf(v16bf a, v16bf b, v8f c) {
    return __builtin_amdgcn_wmma_f32_16x16x32_bf16(false, a, false, b, (short)0, c, false, false);
}

// ---------------------------------------------------------------- utility
__global__ void k_zero(float* p, int n) {
    int i = blockIdx.x * blockDim.x + threadIdx.x;
    if (i < n) p[i] = 0.0f;
}
__global__ void k_f2bf(const float* __restrict__ s, unsigned short* __restrict__ d, int n) {
    int i = blockIdx.x * blockDim.x + threadIdx.x;
    if (i < n) d[i] = f2bf(s[i]);
}
// wt[s][o][k] = w[s][k][o]   (STEPS x H x H)
__global__ void k_ggc_transpose(const float* __restrict__ w, unsigned short* __restrict__ wt) {
    int i = blockIdx.x * blockDim.x + threadIdx.x;
    if (i >= STEPS * HID * HID) return;
    int s = i >> 14, rem = i & 16383, o = rem >> 7, k = rem & 127;
    wt[i] = f2bf(w[(s << 14) + (k << 7) + o]);
}

// ---------------------------------------------------------------- GEMM core (K=128, Ocols=128)
// Persistent waves: wave owns a 64-col group, keeps the full B panel (16 frags)
// in registers, grid-strides over 16-row tiles: 16 back-to-back WMMAs per tile.
#define GEMM_BLOCKS 320   // 2560 waves -> 1280 per column group

// variant 1: fp32 output + bias (input projection)
__global__ void __launch_bounds__(256) k_gemm_f32out(
        const unsigned short* __restrict__ A, const unsigned short* __restrict__ Wt,
        const float* __restrict__ bias, float* __restrict__ outF, int M) {
    int wave = (blockIdx.x * blockDim.x + threadIdx.x) >> 5;
    int lane = threadIdx.x & 31;
    int cg = wave & 1;
    int widx = wave >> 1;
    const int rtStep = (GEMM_BLOCKS * 8) >> 1;
    int rowTiles = M >> 4;
    int c0 = cg << 6;
    int c = lane & 15, hi = lane >> 4;

    v16bf B[4][4];
#pragma unroll
    for (int kc = 0; kc < 4; ++kc)
#pragma unroll
        for (int j = 0; j < 4; ++j)
            B[kc][j] = load_b_frag(Wt, c0 + j * 16, kc * 32, lane);

    float bv[4];
#pragma unroll
    for (int j = 0; j < 4; ++j) bv[j] = bias[c0 + j * 16 + c];

    for (int rt = widx; rt < rowTiles; rt += rtStep) {
        int row0 = rt << 4;
        if (rt + rtStep < rowTiles)
            __builtin_prefetch(A + ((rt + rtStep) * 16 + (lane & 15)) * 128, 0, 3);
        v16bf Af[4];
#pragma unroll
        for (int kc = 0; kc < 4; ++kc) Af[kc] = load_a_frag(A, row0, kc * 32, lane);
        v8f acc[4] = {};
#pragma unroll
        for (int kc = 0; kc < 4; ++kc)
#pragma unroll
            for (int j = 0; j < 4; ++j)
                acc[j] = wmma_bf(Af[kc], B[kc][j], acc[j]);
#pragma unroll
        for (int j = 0; j < 4; ++j) {
            int col = c0 + j * 16 + c;
#pragma unroll
            for (int v = 0; v < 8; ++v)
                outF[(row0 + v + hi * 8) * 128 + col] = acc[j][v] + bv[j];
        }
    }
}

// variant 2: bf16 output, no bias (per-step message GEMM)
__global__ void __launch_bounds__(256) k_gemm_bf16out(
        const unsigned short* __restrict__ A, const unsigned short* __restrict__ Wt,
        unsigned short* __restrict__ outBF, int M) {
    int wave = (blockIdx.x * blockDim.x + threadIdx.x) >> 5;
    int lane = threadIdx.x & 31;
    int cg = wave & 1;
    int widx = wave >> 1;
    const int rtStep = (GEMM_BLOCKS * 8) >> 1;
    int rowTiles = M >> 4;
    int c0 = cg << 6;
    int c = lane & 15, hi = lane >> 4;

    v16bf B[4][4];
#pragma unroll
    for (int kc = 0; kc < 4; ++kc)
#pragma unroll
        for (int j = 0; j < 4; ++j)
            B[kc][j] = load_b_frag(Wt, c0 + j * 16, kc * 32, lane);

    for (int rt = widx; rt < rowTiles; rt += rtStep) {
        int row0 = rt << 4;
        if (rt + rtStep < rowTiles)
            __builtin_prefetch(A + ((rt + rtStep) * 16 + (lane & 15)) * 128, 0, 3);
        v16bf Af[4];
#pragma unroll
        for (int kc = 0; kc < 4; ++kc) Af[kc] = load_a_frag(A, row0, kc * 32, lane);
        v8f acc[4] = {};
#pragma unroll
        for (int kc = 0; kc < 4; ++kc)
#pragma unroll
            for (int j = 0; j < 4; ++j)
                acc[j] = wmma_bf(Af[kc], B[kc][j], acc[j]);
#pragma unroll
        for (int j = 0; j < 4; ++j) {
            int col = c0 + j * 16 + c;
#pragma unroll
            for (int v = 0; v < 8; ++v)
                outBF[(row0 + v + hi * 8) * 128 + col] = f2bf(acc[j][v]);
        }
    }
}

// ---------------------------------------------------------------- column stats (sum / sumsq per column)
__global__ void k_colstats(const float* __restrict__ X, float* __restrict__ st, int M) {
    int c = threadIdx.x;                  // blockDim = 128
    int r0 = blockIdx.x * 256;
    int r1 = r0 + 256; if (r1 > M) r1 = M;
    float s = 0.f, s2 = 0.f;
    for (int r = r0; r < r1; ++r) {
        float v = X[r * 128 + c];
        s += v; s2 += v * v;
    }
    atomicAdd(&st[c], s);
    atomicAdd(&st[128 + c], s2);
}
__global__ void k_bnfin(const float* __restrict__ st, const float* __restrict__ g,
                        const float* __restrict__ b, float* __restrict__ scale,
                        float* __restrict__ shift, float invM) {
    int c = threadIdx.x;                  // 128 threads, 1 block
    float mean = st[c] * invM;
    float var  = st[128 + c] * invM - mean * mean;
    float rs   = rsqrtf(var + 1e-5f);
    float sc   = g[c] * rs;
    scale[c] = sc;
    shift[c] = b[c] - mean * sc;
}
__global__ void k_bn1_apply(const float* __restrict__ pre, const float* __restrict__ scale,
                            const float* __restrict__ shift, float* __restrict__ h,
                            unsigned short* __restrict__ hbf, float* __restrict__ skip) {
    int i = blockIdx.x * blockDim.x + threadIdx.x;
    if (i >= N_NODES * HID) return;
    int c = i & 127;
    float v = pre[i] * scale[c] + shift[c];
    v = v > 0.f ? v : 0.f;
    h[i] = v; hbf[i] = f2bf(v); skip[i] = v;
}

// ---------------------------------------------------------------- edge scatter: agg[dst] += m[src]
// 32 lanes per edge, 4 columns per lane (bf16 gather, fp32 atomic add).
__global__ void k_scatter(const unsigned short* __restrict__ m, const int* __restrict__ src,
                          const int* __restrict__ dst, float* __restrict__ agg) {
    int gid = blockIdx.x * blockDim.x + threadIdx.x;
    int e = gid >> 5;
    if (e >= N_EDGES) return;
    int c4 = (gid & 31) << 2;
    int s = src[e], d = dst[e];
    uint2 raw = *(const uint2*)(m + s * 128 + c4);
    float* q = agg + d * 128 + c4;
    atomicAdd(q + 0, bf2f(raw.x & 0xFFFFu));
    atomicAdd(q + 1, bf2f(raw.x >> 16));
    atomicAdd(q + 2, bf2f(raw.y & 0xFFFFu));
    atomicAdd(q + 3, bf2f(raw.y >> 16));
}

// ---------------------------------------------------------------- fused GRU cell
// Persistent waves: wave owns one 16-col tile, keeps all 6 gate B panels
// (24 frags) in registers, grid-strides over row tiles: 24 WMMAs back-to-back.
#define GRU_BLOCKS 400    // 3200 waves -> 400 per column tile
__global__ void __launch_bounds__(256) k_gru(
        const unsigned short* __restrict__ aggbf, const unsigned short* __restrict__ hbf,
        const unsigned short* __restrict__ wih, const unsigned short* __restrict__ whh,
        const float* __restrict__ bih, const float* __restrict__ bhh,
        float* __restrict__ h, unsigned short* __restrict__ hbf_out) {
    int wave = (blockIdx.x * blockDim.x + threadIdx.x) >> 5;
    int lane = threadIdx.x & 31;
    int ct = wave & 7;
    int widx = wave >> 3;
    const int rtStep = (GRU_BLOCKS * 8) >> 3;
    const int rowTiles = N_NODES / 16;
    int c0 = ct << 4;
    int c = lane & 15, hi = lane >> 4;
    int col = c0 + c;

    // resident gate weight panels: [kchunk][gate]  (0..2 = wih r/z/n, 3..5 = whh r/z/n)
    v16bf B[4][6];
#pragma unroll
    for (int kc = 0; kc < 4; ++kc) {
        int k0 = kc * 32;
        B[kc][0] = load_b_frag(wih, c0,       k0, lane);
        B[kc][1] = load_b_frag(wih, c0 + 128, k0, lane);
        B[kc][2] = load_b_frag(wih, c0 + 256, k0, lane);
        B[kc][3] = load_b_frag(whh, c0,       k0, lane);
        B[kc][4] = load_b_frag(whh, c0 + 128, k0, lane);
        B[kc][5] = load_b_frag(whh, c0 + 256, k0, lane);
    }
    float bir = bih[col], biz = bih[col + 128], bin = bih[col + 256];
    float bhr = bhh[col], bhz = bhh[col + 128], bhn = bhh[col + 256];

    for (int rt = widx; rt < rowTiles; rt += rtStep) {
        int row0 = rt << 4;
        if (rt + rtStep < rowTiles) {
            __builtin_prefetch(aggbf + ((rt + rtStep) * 16 + (lane & 15)) * 128, 0, 3);
            __builtin_prefetch(hbf   + ((rt + rtStep) * 16 + (lane & 15)) * 128, 0, 3);
        }
        v16bf Aa[4], Ah[4];
#pragma unroll
        for (int kc = 0; kc < 4; ++kc) {
            Aa[kc] = load_a_frag(aggbf, row0, kc * 32, lane);
            Ah[kc] = load_a_frag(hbf,   row0, kc * 32, lane);
        }
        v8f aIR = {}, aIZ = {}, aIN = {}, aHR = {}, aHZ = {}, aHN = {};
#pragma unroll
        for (int kc = 0; kc < 4; ++kc) {
            aIR = wmma_bf(Aa[kc], B[kc][0], aIR);
            aIZ = wmma_bf(Aa[kc], B[kc][1], aIZ);
            aIN = wmma_bf(Aa[kc], B[kc][2], aIN);
            aHR = wmma_bf(Ah[kc], B[kc][3], aHR);
            aHZ = wmma_bf(Ah[kc], B[kc][4], aHZ);
            aHN = wmma_bf(Ah[kc], B[kc][5], aHN);
        }
#pragma unroll
        for (int v = 0; v < 8; ++v) {
            int idx = (row0 + v + hi * 8) * 128 + col;
            float hold = h[idx];
            float r  = sigmoidf_((aIR[v] + bir) + (aHR[v] + bhr));
            float z  = sigmoidf_((aIZ[v] + biz) + (aHZ[v] + bhz));
            float nn = tanhf((aIN[v] + bin) + r * (aHN[v] + bhn));
            float hn = (1.0f - z) * nn + z * hold;
            h[idx] = hn;
            hbf_out[idx] = f2bf(hn);
        }
    }
}

// ---------------------------------------------------------------- BN2 + skip + relu + mean-pool accumulate
__global__ void k_bn2_pool(const float* __restrict__ hcur, const float* __restrict__ skip,
                           const float* __restrict__ scale, const float* __restrict__ shift,
                           const int* __restrict__ batch, float* __restrict__ pooled,
                           float* __restrict__ cnt) {
    int i = blockIdx.x * blockDim.x + threadIdx.x;
    if (i >= N_NODES * HID) return;
    int c = i & 127, n = i >> 7;
    float v = hcur[i] * scale[c] + shift[c] + skip[i];
    v = v > 0.f ? v : 0.f;
    int g = batch[n];
    atomicAdd(&pooled[g * 128 + c], v);
    if (c == 0) atomicAdd(&cnt[g], 1.0f);
}

// ---------------------------------------------------------------- head
__global__ void k_head1(const float* __restrict__ pooled, const float* __restrict__ cnt,
                        const float* __restrict__ W1, const float* __restrict__ b1,
                        float* __restrict__ y1, float* __restrict__ st3) {
    int i = blockIdx.x * blockDim.x + threadIdx.x;
    if (i >= N_GRAPH * HID) return;
    int g = i >> 7, c = i & 127;
    float inv = 1.0f / fmaxf(cnt[g], 1.0f);
    const float* mrow = pooled + g * 128;
    const float* w = W1 + c * 256;
    float acc = b1[c];
#pragma unroll 4
    for (int k = 0; k < 128; ++k) acc += mrow[k] * inv * (w[k] + w[128 + k]);
    y1[i] = acc;
    atomicAdd(&st3[c], acc);
    atomicAdd(&st3[128 + c], acc * acc);
}
__global__ void k_head2(const float* __restrict__ y1, const float* __restrict__ scale,
                        const float* __restrict__ shift, const float* __restrict__ W2,
                        const float* __restrict__ b2, float* __restrict__ out) {
    int i = blockIdx.x * blockDim.x + threadIdx.x;
    if (i >= N_GRAPH * OUTD) return;
    int g = i >> 1, o = i & 1;
    const float* row = y1 + g * 128;
    const float* w = W2 + o * 128;
    float acc = b2[o];
#pragma unroll 4
    for (int c = 0; c < 128; ++c) {
        float v = row[c] * scale[c] + shift[c];
        v = v > 0.f ? v : 0.f;
        acc += v * w[c];
    }
    out[i] = acc;
}

// ---------------------------------------------------------------- launch
extern "C" void kernel_launch(void* const* d_in, const int* in_sizes, int n_in,
                              void* d_out, int out_size, void* d_ws, size_t ws_size,
                              hipStream_t stream) {
    (void)in_sizes; (void)n_in; (void)out_size; (void)ws_size;
    const float* x       = (const float*)d_in[0];
    const int*   edge    = (const int*)d_in[1];     // [2,E]: src then dst
    const int*   batch   = (const int*)d_in[2];
    const float* W_in    = (const float*)d_in[3];
    const float* b_in    = (const float*)d_in[4];
    const float* bn1_g   = (const float*)d_in[5];
    const float* bn1_b   = (const float*)d_in[6];
    const float* ggc_w   = (const float*)d_in[7];
    const float* gru_wih = (const float*)d_in[8];
    const float* gru_whh = (const float*)d_in[9];
    const float* gru_bih = (const float*)d_in[10];
    const float* gru_bhh = (const float*)d_in[11];
    const float* bn2_g   = (const float*)d_in[12];
    const float* bn2_b   = (const float*)d_in[13];
    const float* W1      = (const float*)d_in[14];
    const float* b1      = (const float*)d_in[15];
    const float* bn3_g   = (const float*)d_in[16];
    const float* bn3_b   = (const float*)d_in[17];
    const float* W2      = (const float*)d_in[18];
    const float* b2      = (const float*)d_in[19];

    char* ws = (char*)d_ws;
    size_t off = 0;
    auto alloc = [&](size_t bytes) -> void* {
        void* p = ws + off;
        off = (off + bytes + 255) & ~(size_t)255;
        return p;
    };
    const size_t NH = (size_t)N_NODES * HID;
    float*          h      = (float*)alloc(NH * 4);
    float*          skip   = (float*)alloc(NH * 4);
    float*          agg    = (float*)alloc(NH * 4);          // also GEMM1 pre-BN output
    unsigned short* hbfA   = (unsigned short*)alloc(NH * 2); // also x_bf (dead after GEMM1)
    unsigned short* hbfB   = (unsigned short*)alloc(NH * 2);
    unsigned short* mbf    = (unsigned short*)alloc(NH * 2); // also agg_bf (m dead after scatter)
    unsigned short* winbf  = (unsigned short*)alloc(HID * DIM * 2);
    unsigned short* ggcwT  = (unsigned short*)alloc(STEPS * HID * HID * 2);
    unsigned short* wihbf  = (unsigned short*)alloc(3 * HID * HID * 2);
    unsigned short* whhbf  = (unsigned short*)alloc(3 * HID * HID * 2);
    float*          st     = (float*)alloc(2 * HID * 4);
    float*          scale  = (float*)alloc(HID * 4);
    float*          shift  = (float*)alloc(HID * 4);
    float*          pooled = (float*)alloc(N_GRAPH * HID * 4);
    float*          cnt    = (float*)alloc(N_GRAPH * 4);
    float*          y1     = (float*)alloc(N_GRAPH * HID * 4);
    float*          st3    = (float*)alloc(2 * HID * 4);
    float*          scale3 = (float*)alloc(HID * 4);
    float*          shift3 = (float*)alloc(HID * 4);

    auto cdiv = [](int a, int b) { return (a + b - 1) / b; };
    const int elemN = N_NODES * HID;                    // 12.8M

    // ---- prep: bf16 conversions / transposes
    k_f2bf<<<cdiv(elemN, 256), 256, 0, stream>>>(x, hbfA, elemN);               // x_bf
    k_f2bf<<<cdiv(HID * DIM, 256), 256, 0, stream>>>(W_in, winbf, HID * DIM);
    k_ggc_transpose<<<cdiv(STEPS * HID * HID, 256), 256, 0, stream>>>(ggc_w, ggcwT);
    k_f2bf<<<cdiv(3 * HID * HID, 256), 256, 0, stream>>>(gru_wih, wihbf, 3 * HID * HID);
    k_f2bf<<<cdiv(3 * HID * HID, 256), 256, 0, stream>>>(gru_whh, whhbf, 3 * HID * HID);

    // ---- input projection + BN1 + relu
    k_gemm_f32out<<<GEMM_BLOCKS, 256, 0, stream>>>(hbfA, winbf, b_in, agg, N_NODES);
    k_zero<<<1, 256, 0, stream>>>(st, 2 * HID);
    k_colstats<<<cdiv(N_NODES, 256), 128, 0, stream>>>(agg, st, N_NODES);
    k_bnfin<<<1, 128, 0, stream>>>(st, bn1_g, bn1_b, scale, shift, 1.0f / N_NODES);
    k_bn1_apply<<<cdiv(elemN, 256), 256, 0, stream>>>(agg, scale, shift, h, hbfA, skip);

    // ---- gated graph conv steps
    const int* src = edge;
    const int* dst = edge + N_EDGES;
    for (int s = 0; s < STEPS; ++s) {
        unsigned short* cur = (s & 1) ? hbfB : hbfA;
        unsigned short* nxt = (s & 1) ? hbfA : hbfB;
        k_gemm_bf16out<<<GEMM_BLOCKS, 256, 0, stream>>>(cur, ggcwT + s * HID * HID, mbf, N_NODES);
        k_zero<<<cdiv(elemN, 256), 256, 0, stream>>>(agg, elemN);
        k_scatter<<<cdiv(N_EDGES * 32, 256), 256, 0, stream>>>(mbf, src, dst, agg);
        k_f2bf<<<cdiv(elemN, 256), 256, 0, stream>>>(agg, mbf, elemN);          // agg -> bf16
        k_gru<<<GRU_BLOCKS, 256, 0, stream>>>(mbf, cur, wihbf, whhbf,
                                              gru_bih, gru_bhh, h, nxt);
    }

    // ---- BN2 + skip + relu + mean pool
    k_zero<<<1, 256, 0, stream>>>(st, 2 * HID);
    k_colstats<<<cdiv(N_NODES, 256), 128, 0, stream>>>(h, st, N_NODES);
    k_bnfin<<<1, 128, 0, stream>>>(st, bn2_g, bn2_b, scale, shift, 1.0f / N_NODES);
    k_zero<<<cdiv(N_GRAPH * HID, 256), 256, 0, stream>>>(pooled, N_GRAPH * HID);
    k_zero<<<1, 256, 0, stream>>>(cnt, N_GRAPH);
    k_bn2_pool<<<cdiv(elemN, 256), 256, 0, stream>>>(h, skip, scale, shift, batch, pooled, cnt);

    // ---- head
    k_zero<<<1, 256, 0, stream>>>(st3, 2 * HID);
    k_head1<<<cdiv(N_GRAPH * HID, 256), 256, 0, stream>>>(pooled, cnt, W1, b1, y1, st3);
    k_bnfin<<<1, 128, 0, stream>>>(st3, bn3_g, bn3_b, scale3, shift3, 1.0f / N_GRAPH);
    k_head2<<<cdiv(N_GRAPH * OUTD, 256), 256, 0, stream>>>(y1, scale3, shift3, W2, b2, (float*)d_out);
}
